// my_attention_head_10299331576605
// MI455X (gfx1250) — matrix operand order repaired and checked by reading, not compiled, not run on hardware
//
#include <hip/hip_runtime.h>
#include <hip/hip_bf16.h>
#include <math.h>

#define D_MODEL 1024
#define D_K     256
#define SEQ     4096
#define BATCH   4
#define NTOT    (BATCH * SEQ)   // 16384 rows total

typedef __attribute__((ext_vector_type(16))) __bf16         bf16x16;
typedef __attribute__((ext_vector_type(8)))  float          f32x8;
typedef __attribute__((ext_vector_type(8)))  unsigned short u16x8;

static __device__ __forceinline__ unsigned short bf16_bits(float x) {
  __bf16 v = (__bf16)x;
  return __builtin_bit_cast(unsigned short, v);
}

// ---- fragment loaders ------------------------------------------------------
// A-matrix 16x32 (bf16): lane = m + 16h holds row m;
// elements e=0..7 -> K = h*8 + e ; e=8..15 -> K = 16 + h*8 + (e-8)
static __device__ __forceinline__ bf16x16 a_frag_from_f32_row(const float* row, int kbase, int h) {
  bf16x16 f;
  const float* p0 = row + kbase + h * 8;
  const float* p1 = row + kbase + 16 + h * 8;
#pragma unroll
  for (int i = 0; i < 8; ++i) { f[i] = (__bf16)p0[i]; f[i + 8] = (__bf16)p1[i]; }
  return f;
}
static __device__ __forceinline__ bf16x16 a_frag_from_bf16_row(const __bf16* row, int kbase, int h) {
  bf16x16 f;
  const __bf16* p0 = row + kbase + h * 8;
  const __bf16* p1 = row + kbase + 16 + h * 8;
#pragma unroll
  for (int i = 0; i < 8; ++i) { f[i] = p0[i]; f[i + 8] = p1[i]; }
  return f;
}
// B-matrix 32x16: lane = n + 16h holds column n;
// elements e=0..7 -> K = h*8 + e ; e=8..15 -> K = 16 + h*8 + (e-8)
static __device__ __forceinline__ bf16x16 b_frag_from_bf16_row(const __bf16* row, int kbase, int h) {
  bf16x16 f;
  const __bf16* p0 = row + kbase + h * 8;
  const __bf16* p1 = row + kbase + 16 + h * 8;
#pragma unroll
  for (int i = 0; i < 8; ++i) { f[i] = p0[i]; f[i + 8] = p1[i]; }
  return f;
}

// ---- kernel 0: one-time f32 -> bf16 conversion (weights) -------------------
__global__ void f32_to_bf16_kernel(const float* __restrict__ src,
                                   unsigned short* __restrict__ dst, int nelem) {
  int i = blockIdx.x * 256 + threadIdx.x;
  if (i < nelem) dst[i] = bf16_bits(src[i]);
}

// ---- kernel 1: projection --------------------------------------------------
// One wave computes a full 16-row x 256-col block: X read exactly once.
// B fragments (and the A fragment across k-chunks) are double-buffered so the
// next tile's global_load_b128 issues before the current v_wmma.
__global__ __launch_bounds__(32) void proj_bf16_kernel(const float* __restrict__ X,
                                                       const unsigned short* __restrict__ W_us,
                                                       const float* __restrict__ bias,
                                                       unsigned short* __restrict__ dst_us,
                                                       int transposed) {
  const __bf16* Wbf = (const __bf16*)W_us;
  __bf16* dst = (__bf16*)dst_us;
  const int lane = threadIdx.x & 31;
  const int m = lane & 15;   // A row / B column within tile
  const int h = lane >> 4;
  const int row0 = blockIdx.x * 16;   // over N = B*S

  const float* xrow = X + (size_t)(row0 + m) * D_MODEL;
  const __bf16* wcol = Wbf + (size_t)m * D_MODEL;  // + t*16*D_MODEL selects tile
  f32x8 zero = {};
  f32x8 oacc[16];
#pragma unroll
  for (int t = 0; t < 16; ++t) oacc[t] = zero;

  bf16x16 a = a_frag_from_f32_row(xrow, 0, h);
  bf16x16 b = b_frag_from_bf16_row(wcol, 0, h);
  for (int c = 0; c < D_MODEL / 32; ++c) {
    bf16x16 anext = a;
    if (c + 1 < D_MODEL / 32) anext = a_frag_from_f32_row(xrow, (c + 1) * 32, h);
#pragma unroll
    for (int t = 0; t < 16; ++t) {
      bf16x16 bnext = b;
      if (t + 1 < 16)
        bnext = b_frag_from_bf16_row(wcol + (size_t)(t + 1) * 16 * D_MODEL, c * 32, h);
      else if (c + 1 < D_MODEL / 32)
        bnext = b_frag_from_bf16_row(wcol, (c + 1) * 32, h);
      oacc[t] = __builtin_amdgcn_wmma_f32_16x16x32_bf16(false, a, false, b, (short)0, oacc[t], false, false);
      b = bnext;
    }
    a = anext;
  }

#pragma unroll
  for (int t = 0; t < 16; ++t) {
    const float bv = bias[t * 16 + m];
    if (!transposed) {
#pragma unroll
      for (int r = 0; r < 8; ++r)
        dst[(size_t)(row0 + r + 8 * h) * D_K + t * 16 + m] = (__bf16)(oacc[t][r] + bv);
    } else {
      u16x8 pk;
#pragma unroll
      for (int r = 0; r < 8; ++r) pk[r] = bf16_bits(oacc[t][r] + bv);
      // column index (t*16+m), 8 consecutive rows starting at row0 + 8h
      *(u16x8*)(dst + (size_t)(t * 16 + m) * NTOT + row0 + 8 * h) = pk;
    }
  }
}

// ---- kernel 2: flash attention, one wave per 16-query block ----------------
__global__ __launch_bounds__(32) void flash_attn_kernel(const unsigned short* __restrict__ qus,
                                                        const unsigned short* __restrict__ kus,
                                                        const unsigned short* __restrict__ vTus,
                                                        float* __restrict__ out) {
  __shared__ __bf16 pLDS[16 * 32];
  const __bf16* qb = (const __bf16*)qus;
  const __bf16* kb = (const __bf16*)kus;
  const __bf16* vT = (const __bf16*)vTus;   // [D_K][NTOT]

  const int lane = threadIdx.x & 31;
  const int n = lane & 15;
  const int h = lane >> 4;
  const int bidx = blockIdx.x >> 8;          // SEQ/16 = 256 query tiles per batch
  const int q0 = (blockIdx.x & 255) * 16;

  // Q block 16 x 256 as 8 A-fragments (K chunks of 32)
  bf16x16 qf[8];
  const __bf16* qrow = qb + ((size_t)bidx * SEQ + q0 + n) * D_K;
#pragma unroll
  for (int c = 0; c < 8; ++c) qf[c] = a_frag_from_bf16_row(qrow, c * 32, h);

  f32x8 zero = {};
  f32x8 oacc[16];
#pragma unroll
  for (int t = 0; t < 16; ++t) oacc[t] = zero;
  float mrow[8], lrow[8];
#pragma unroll
  for (int r = 0; r < 8; ++r) { mrow[r] = -1e30f; lrow[r] = 0.0f; }

  const __bf16* kbB = kb + (size_t)bidx * SEQ * D_K;
  const __bf16* vTB = vT + (size_t)bidx * SEQ;     // + (d)*NTOT + s
  const float scale = 0.0625f;  // 1/sqrt(256)

  for (int s0 = 0; s0 < SEQ; s0 += 32) {
    if (s0 + 32 < SEQ) {                      // CDNA5 global_prefetch_b8
      __builtin_prefetch(kbB + (size_t)(s0 + 32 + n) * D_K, 0, 1);
      __builtin_prefetch(vTB + (size_t)n * NTOT + s0 + 32, 0, 1);
    }
    // scores S = Q . K^T : two 16x16 f32 tiles (32 keys), double-buffered B
    f32x8 sacc0 = zero, sacc1 = zero;
    {
      const __bf16* krow0 = kbB + (size_t)(s0 + n) * D_K;
      const __bf16* krow1 = kbB + (size_t)(s0 + 16 + n) * D_K;
      bf16x16 b0 = b_frag_from_bf16_row(krow0, 0, h);
      bf16x16 b1 = b_frag_from_bf16_row(krow1, 0, h);
#pragma unroll
      for (int c = 0; c < 8; ++c) {
        bf16x16 n0 = b0, n1 = b1;
        if (c + 1 < 8) {
          n0 = b_frag_from_bf16_row(krow0, (c + 1) * 32, h);
          n1 = b_frag_from_bf16_row(krow1, (c + 1) * 32, h);
        }
        sacc0 = __builtin_amdgcn_wmma_f32_16x16x32_bf16(false, qf[c], false, b0, (short)0, sacc0, false, false);
        sacc1 = __builtin_amdgcn_wmma_f32_16x16x32_bf16(false, qf[c], false, b1, (short)0, sacc1, false, false);
        b0 = n0; b1 = n1;
      }
    }
    // online softmax update (row = r + 8h; reduce across 16-lane half)
    float corr[8];
#pragma unroll
    for (int r = 0; r < 8; ++r) {
      float s0v = sacc0[r] * scale;
      float s1v = sacc1[r] * scale;
      float vmax = fmaxf(s0v, s1v);
      vmax = fmaxf(vmax, __shfl_xor(vmax, 1, 32));
      vmax = fmaxf(vmax, __shfl_xor(vmax, 2, 32));
      vmax = fmaxf(vmax, __shfl_xor(vmax, 4, 32));
      vmax = fmaxf(vmax, __shfl_xor(vmax, 8, 32));
      float mnew = fmaxf(mrow[r], vmax);
      float cr = __expf(mrow[r] - mnew);
      float p0 = __expf(s0v - mnew);
      float p1 = __expf(s1v - mnew);
      float rs = p0 + p1;
      rs += __shfl_xor(rs, 1, 32);
      rs += __shfl_xor(rs, 2, 32);
      rs += __shfl_xor(rs, 4, 32);
      rs += __shfl_xor(rs, 8, 32);
      lrow[r] = lrow[r] * cr + rs;
      mrow[r] = mnew;
      corr[r] = cr;
      pLDS[(r + 8 * h) * 32 + n]      = (__bf16)p0;   // stage P tile (16x32)
      pLDS[(r + 8 * h) * 32 + 16 + n] = (__bf16)p1;
    }
#pragma unroll
    for (int t = 0; t < 16; ++t)
#pragma unroll
      for (int r = 0; r < 8; ++r) oacc[t][r] *= corr[r];

    asm volatile("s_wait_dscnt 0" ::: "memory");  // LDS RAW fence (same wave)

    // reload P in A-fragment layout; O += P . V  (V transposed: contiguous,
    // double-buffered B fragments)
    bf16x16 pf = a_frag_from_bf16_row(&pLDS[n * 32], 0, h);
    {
      const __bf16* vrow = vTB + (size_t)n * NTOT;   // + t*16*NTOT selects tile
      bf16x16 vb = b_frag_from_bf16_row(vrow, s0, h);
#pragma unroll
      for (int t = 0; t < 16; ++t) {
        bf16x16 vnext = vb;
        if (t + 1 < 16)
          vnext = b_frag_from_bf16_row(vrow + (size_t)(t + 1) * 16 * NTOT, s0, h);
        oacc[t] = __builtin_amdgcn_wmma_f32_16x16x32_bf16(false, pf, false, vb, (short)0, oacc[t], false, false);
        vb = vnext;
      }
    }
  }

  // epilogue: normalize by l and store f32
  float* orow = out + ((size_t)bidx * SEQ + q0) * D_K;
#pragma unroll
  for (int t = 0; t < 16; ++t)
#pragma unroll
    for (int r = 0; r < 8; ++r) {
      float v = oacc[t][r] / lrow[r];
      orow[(size_t)(r + 8 * h) * D_K + t * 16 + n] = v;
    }
}

// ---- host launch -----------------------------------------------------------
extern "C" void kernel_launch(void* const* d_in, const int* in_sizes, int n_in,
                              void* d_out, int out_size, void* d_ws, size_t ws_size,
                              hipStream_t stream) {
  const float* q  = (const float*)d_in[0];
  const float* k  = (const float*)d_in[1];
  const float* v  = (const float*)d_in[2];
  // d_in[3] = mask (unused by the reference forward)
  const float* Wq = (const float*)d_in[4];
  const float* bq = (const float*)d_in[5];
  const float* Wk = (const float*)d_in[6];
  const float* bk = (const float*)d_in[7];
  const float* Wv = (const float*)d_in[8];
  const float* bv = (const float*)d_in[9];
  float* out = (float*)d_out;

  const size_t qkv_elems = (size_t)NTOT * D_K;       // 4,194,304 bf16 each
  const size_t w_elems   = (size_t)D_K * D_MODEL;    // 262,144 bf16 each
  unsigned short* qb   = (unsigned short*)d_ws;
  unsigned short* kb   = qb + qkv_elems;
  unsigned short* vbT  = kb + qkv_elems;             // stored transposed [D_K][NTOT]
  unsigned short* Wqb  = vbT + qkv_elems;
  unsigned short* Wkb  = Wqb + w_elems;
  unsigned short* Wvb  = Wkb + w_elems;

  // one-time weight conversion to bf16
  {
    dim3 cg((unsigned)((w_elems + 255) / 256));
    f32_to_bf16_kernel<<<cg, 256, 0, stream>>>(Wq, Wqb, (int)w_elems);
    f32_to_bf16_kernel<<<cg, 256, 0, stream>>>(Wk, Wkb, (int)w_elems);
    f32_to_bf16_kernel<<<cg, 256, 0, stream>>>(Wv, Wvb, (int)w_elems);
  }

  dim3 pgrid(NTOT / 16);   // 1024 waves; each produces a full 16x256 block
  proj_bf16_kernel<<<pgrid, 32, 0, stream>>>(q, Wqb, bq, qb, 0);
  proj_bf16_kernel<<<pgrid, 32, 0, stream>>>(k, Wkb, bk, kb, 0);
  proj_bf16_kernel<<<pgrid, 32, 0, stream>>>(v, Wvb, bv, vbT, 1);

  flash_attn_kernel<<<dim3(NTOT / 16), 32, 0, stream>>>(qb, kb, vbT, out);
}
